// GATModel_53128745451599
// MI455X (gfx1250) — compile-verified, matrix operand order checked by reading, not run on hardware
//
#include <hip/hip_runtime.h>
#include <hip/hip_bf16.h>

// ---------------- types for CDNA5 WMMA ----------------
typedef __attribute__((ext_vector_type(16))) __bf16   v16bf;
typedef __attribute__((ext_vector_type(8)))  float    v8f;
typedef __attribute__((ext_vector_type(8)))  unsigned v8u;

#define NEG_SLOPE 0.2f

// ---------------- small device helpers ----------------
__device__ __forceinline__ unsigned short f2bf(float f) {
    unsigned u = __float_as_uint(f);
    unsigned r = (u + 0x7FFFu + ((u >> 16) & 1u)) >> 16;   // round-to-nearest-even
    return (unsigned short)r;
}
// monotone bijection float -> uint (order preserving), for atomicMax-based segment max
__device__ __forceinline__ unsigned ordf(float f) {
    unsigned u = __float_as_uint(f);
    return (u & 0x80000000u) ? ~u : (u | 0x80000000u);
}
__device__ __forceinline__ float unordf(unsigned u) {
    return (u & 0x80000000u) ? __uint_as_float(u & 0x7FFFFFFFu) : __uint_as_float(~u);
}

// ---------------- utility kernels ----------------
__global__ void zero_u32(unsigned* __restrict__ p, long long n) {
    long long i = (long long)blockIdx.x * blockDim.x + threadIdx.x;
    if (i < n) p[i] = 0u;
}

__global__ void conv_bf(const float* __restrict__ in, unsigned short* __restrict__ out, int total) {
    int i = blockIdx.x * blockDim.x + threadIdx.x;
    if (i < total) out[i] = f2bf(in[i]);
}

// W [K, Nn] row-major f32  ->  Wt [Nn, K] row-major bf16 (so B-fragment pairs are contiguous)
__global__ void tconv_bf(const float* __restrict__ W, unsigned short* __restrict__ Wt, int K, int Nn) {
    int i = blockIdx.x * blockDim.x + threadIdx.x;
    if (i >= K * Nn) return;
    int n = i / K, k = i % K;
    Wt[i] = f2bf(W[(size_t)k * Nn + n]);
}

// ---------------- bf16 WMMA GEMM: C[M,Nn] = A[M,K] * B[K,Nn] (B given transposed [Nn,K]) ----------
// One wave computes one 16x16 tile of C. K must be a multiple of 32, Nn a multiple of 16.
__global__ void gemm_bf16_wmma(const unsigned short* __restrict__ A,
                               const unsigned short* __restrict__ Bt,
                               float* __restrict__ C,
                               int M, int Nn, int K,
                               int tilesPerRow, int totalTiles)
{
    int wave = blockIdx.x * (blockDim.x >> 5) + (threadIdx.x >> 5);
    if (wave >= totalTiles) return;      // uniform per wave -> EXEC all ones for WMMA
    int mt = wave / tilesPerRow;
    int nt = wave % tilesPerRow;
    int lane = threadIdx.x & 31;
    int row  = lane & 15;                // M (for A) / N (for B) index within tile
    int kg   = lane >> 4;                // K-group select (lanes 16-31 hold K+8 / K+24)

    int arow = mt * 16 + row; if (arow > M - 1) arow = M - 1;   // clamp (M=50000 divides 16, safety only)
    const unsigned short* ap = A  + (size_t)arow * K;
    const unsigned short* bp = Bt + (size_t)(nt * 16 + row) * K;

    v8f acc = {0.f, 0.f, 0.f, 0.f, 0.f, 0.f, 0.f, 0.f};

    for (int kb = 0; kb < K; kb += 32) {
        v8u au, bu;
#pragma unroll
        for (int i = 0; i < 8; ++i) {
            // 16-bit A 16x32 layout: VGPR i<4 -> K = kg*8 + 2i ; i>=4 -> K = 16 + kg*8 + 2(i-4)
            int k = kb + ((i >> 2) << 4) + (kg << 3) + ((i & 3) << 1);
            au[i] = *(const unsigned*)(ap + k);   // contiguous bf16 pair
            bu[i] = *(const unsigned*)(bp + k);   // contiguous bf16 pair (Bt is [Nn,K])
        }
        v16bf a = __builtin_bit_cast(v16bf, au);
        v16bf b = __builtin_bit_cast(v16bf, bu);
        acc = __builtin_amdgcn_wmma_f32_16x16x32_bf16(
            /*neg_a=*/false, a, /*neg_b=*/false, b,
            /*c_mod=*/(short)0, acc, /*reuse_a=*/false, /*reuse_b=*/false);
    }

    // C/D layout: VGPR r -> M = r + 8*kg ; N = lane&15
    int n = nt * 16 + (lane & 15);
#pragma unroll
    for (int r = 0; r < 8; ++r) {
        int m = mt * 16 + r + (kg << 3);
        if (m < M) C[(size_t)m * Nn + n] = acc[r];
    }
}

// ---------------- attention dot products: alpha_s/d[n,h] = sum_c h[n,h,c]*a[h,c] --------------
__global__ void attn_dot(const float* __restrict__ h,
                         const float* __restrict__ asrc, const float* __restrict__ adst,
                         float* __restrict__ outs, float* __restrict__ outd,
                         int N, int H, int Cc)
{
    int idx = blockIdx.x * blockDim.x + threadIdx.x;
    if (idx >= N * H) return;
    int n = idx / H, hh = idx % H;
    const float* hp = h + (size_t)n * H * Cc + (size_t)hh * Cc;
    const float* sp = asrc + (size_t)hh * Cc;
    const float* dp = adst + (size_t)hh * Cc;
    float s = 0.f, d = 0.f;
    for (int c = 0; c < Cc; ++c) { float v = hp[c]; s += v * sp[c]; d += v * dp[c]; }
    outs[idx] = s;
    outd[idx] = d;
}

// ---------------- edge pass 1: segment max of leaky_relu(as[src]+ad[dst]) over dst ------------
__global__ void edge_max(const float* __restrict__ as, const float* __restrict__ ad,
                         const int* __restrict__ src, const int* __restrict__ dst,
                         unsigned* __restrict__ emax, int E, int H)
{
    long long idx = (long long)blockIdx.x * blockDim.x + threadIdx.x;
    if (idx >= (long long)E * H) return;
    int e = (int)(idx / H), hh = (int)(idx % H);
    int s = src[e], d = dst[e];
    float t = as[(size_t)s * H + hh] + ad[(size_t)d * H + hh];
    t = t > 0.f ? t : NEG_SLOPE * t;
    atomicMax(&emax[(size_t)d * H + hh], ordf(t));
}

// ---------------- edge pass 2: e_exp = exp(e - max[dst]); denom[dst] += e_exp -----------------
__global__ void edge_exp(const float* __restrict__ as, const float* __restrict__ ad,
                         const int* __restrict__ src, const int* __restrict__ dst,
                         const unsigned* __restrict__ emax,
                         float* __restrict__ eexp, float* __restrict__ denom, int E, int H)
{
    long long idx = (long long)blockIdx.x * blockDim.x + threadIdx.x;
    if (idx >= (long long)E * H) return;
    int e = (int)(idx / H), hh = (int)(idx % H);
    int s = src[e], d = dst[e];
    float t = as[(size_t)s * H + hh] + ad[(size_t)d * H + hh];
    t = t > 0.f ? t : NEG_SLOPE * t;
    float m  = unordf(emax[(size_t)d * H + hh]);
    float ex = __expf(t - m);
    eexp[idx] = ex;
    atomicAdd(&denom[(size_t)d * H + hh], ex);
}

// ---------------- edge pass 3: out[dst,h,c] += h[src,h,c] * e_exp/denom[dst,h] ----------------
__global__ void edge_scatter(const float* __restrict__ h, const float* __restrict__ eexp,
                             const float* __restrict__ denom,
                             const int* __restrict__ src, const int* __restrict__ dst,
                             float* __restrict__ out, int E, int H, int Cc)
{
    int HC = H * Cc;
    long long idx = (long long)blockIdx.x * blockDim.x + threadIdx.x;
    if (idx >= (long long)E * HC) return;
    int e = (int)(idx / HC), j = (int)(idx % HC);
    int hh = j / Cc;
    int s = src[e], d = dst[e];
    float alpha = eexp[(size_t)e * H + hh] / denom[(size_t)d * H + hh];
    atomicAdd(&out[(size_t)d * HC + j], h[(size_t)s * HC + j] * alpha);
}

// ---------------- bias + ELU, re-quantize to bf16 (layer-2 GEMM input) ------------------------
__global__ void bias_elu_bf(const float* __restrict__ in, const float* __restrict__ bias,
                            unsigned short* __restrict__ outbf, int total, int HC)
{
    int i = blockIdx.x * blockDim.x + threadIdx.x;
    if (i >= total) return;
    float v = in[i] + bias[i % HC];
    v = v > 0.f ? v : (__expf(v) - 1.f);
    outbf[i] = f2bf(v);
}

// ---------------- final bias add ----------------
__global__ void bias_out(const float* __restrict__ in, const float* __restrict__ bias,
                         float* __restrict__ out, int total, int Cc)
{
    int i = blockIdx.x * blockDim.x + threadIdx.x;
    if (i >= total) return;
    out[i] = in[i] + bias[i % Cc];
}

// =====================================================================================
extern "C" void kernel_launch(void* const* d_in, const int* in_sizes, int n_in,
                              void* d_out, int out_size, void* d_ws, size_t ws_size,
                              hipStream_t stream)
{
    const float* x      = (const float*)d_in[0];
    const int*   ei     = (const int*)  d_in[1];   // [2, E]: row0=src, row1=dst
    const float* W1     = (const float*)d_in[2];   // [128, 128]
    const float* a_src1 = (const float*)d_in[3];   // [4, 32]
    const float* a_dst1 = (const float*)d_in[4];
    const float* b1     = (const float*)d_in[5];   // [128]
    const float* W2     = (const float*)d_in[6];   // [128, 64]
    const float* a_src2 = (const float*)d_in[7];   // [1, 64]
    const float* a_dst2 = (const float*)d_in[8];
    const float* b2     = (const float*)d_in[9];   // [64]

    const int IN = 128, H1 = 4, C1 = 32, HC1 = 128, OUT = 64;
    const int N = in_sizes[0] / IN;
    const int E = in_sizes[1] / 2;
    const int* src = ei;
    const int* dst = ei + E;

    // ---- workspace carve-out ----
    char* ws = (char*)d_ws;
    size_t off = 0;
    auto carve = [&](size_t bytes) -> char* {
        char* p = ws + off;
        off = (off + bytes + 255) & ~(size_t)255;
        return p;
    };
    unsigned short* xbf    = (unsigned short*)carve((size_t)N * IN * 2);
    unsigned short* w1t    = (unsigned short*)carve((size_t)HC1 * IN * 2);   // [128 n][128 k]
    unsigned short* w2t    = (unsigned short*)carve((size_t)OUT * HC1 * 2);  // [64 n][128 k]
    float*    h1     = (float*)   carve((size_t)N * HC1 * 4);
    float*    as1    = (float*)   carve((size_t)N * H1 * 4);
    float*    ad1    = (float*)   carve((size_t)N * H1 * 4);
    unsigned* emax1  = (unsigned*)carve((size_t)N * H1 * 4);
    float*    eexp1  = (float*)   carve((size_t)E * H1 * 4);
    float*    den1   = (float*)   carve((size_t)N * H1 * 4);
    float*    agg1   = (float*)   carve((size_t)N * HC1 * 4);
    unsigned short* h1ebf = (unsigned short*)carve((size_t)N * HC1 * 2);
    float*    g2     = (float*)   carve((size_t)N * OUT * 4);
    float*    as2    = (float*)   carve((size_t)N * 4);
    float*    ad2    = (float*)   carve((size_t)N * 4);
    unsigned* emax2  = (unsigned*)carve((size_t)N * 4);
    float*    eexp2  = (float*)   carve((size_t)E * 4);
    float*    den2   = (float*)   carve((size_t)N * 4);
    float*    agg2   = (float*)   carve((size_t)N * OUT * 4);
    (void)ws_size;

    const int T = 256;
    auto blocks = [](long long n, int t) { return (unsigned)((n + t - 1) / t); };

    // ---- zero accumulators (fresh every call; graph-replay safe) ----
    zero_u32<<<blocks((long long)N * HC1, T), T, 0, stream>>>((unsigned*)agg1, (long long)N * HC1);
    zero_u32<<<blocks((long long)N * H1,  T), T, 0, stream>>>(emax1,           (long long)N * H1);
    zero_u32<<<blocks((long long)N * H1,  T), T, 0, stream>>>((unsigned*)den1, (long long)N * H1);
    zero_u32<<<blocks((long long)N * OUT, T), T, 0, stream>>>((unsigned*)agg2, (long long)N * OUT);
    zero_u32<<<blocks((long long)N,       T), T, 0, stream>>>(emax2,           (long long)N);
    zero_u32<<<blocks((long long)N,       T), T, 0, stream>>>((unsigned*)den2, (long long)N);

    // ---- precision prep: f32 -> bf16 (weights pre-transposed for B-fragment loads) ----
    conv_bf <<<blocks((long long)N * IN, T), T, 0, stream>>>(x, xbf, N * IN);
    tconv_bf<<<blocks((long long)IN * HC1, T), T, 0, stream>>>(W1, w1t, IN, HC1);
    tconv_bf<<<blocks((long long)HC1 * OUT, T), T, 0, stream>>>(W2, w2t, HC1, OUT);

    // ================= layer 1 =================
    {   // h1 = x @ W1  : M=N, Nn=128, K=128  (WMMA bf16)
        int mtiles = N / 16, ntiles = HC1 / 16;
        int total = mtiles * ntiles;
        gemm_bf16_wmma<<<blocks(total, 8) , 256, 0, stream>>>(xbf, w1t, h1, N, HC1, IN, ntiles, total);
    }
    attn_dot<<<blocks((long long)N * H1, T), T, 0, stream>>>(h1, a_src1, a_dst1, as1, ad1, N, H1, C1);
    edge_max<<<blocks((long long)E * H1, T), T, 0, stream>>>(as1, ad1, src, dst, emax1, E, H1);
    edge_exp<<<blocks((long long)E * H1, T), T, 0, stream>>>(as1, ad1, src, dst, emax1, eexp1, den1, E, H1);
    edge_scatter<<<blocks((long long)E * HC1, T), T, 0, stream>>>(h1, eexp1, den1, src, dst, agg1, E, H1, C1);
    bias_elu_bf<<<blocks((long long)N * HC1, T), T, 0, stream>>>(agg1, b1, h1ebf, N * HC1, HC1);

    // ================= layer 2 =================
    {   // g2 = elu(h1) @ W2 : M=N, Nn=64, K=128  (WMMA bf16)
        int mtiles = N / 16, ntiles = OUT / 16;
        int total = mtiles * ntiles;
        gemm_bf16_wmma<<<blocks(total, 8), 256, 0, stream>>>(h1ebf, w2t, g2, N, OUT, HC1, ntiles, total);
    }
    attn_dot<<<blocks((long long)N, T), T, 0, stream>>>(g2, a_src2, a_dst2, as2, ad2, N, 1, OUT);
    edge_max<<<blocks((long long)E, T), T, 0, stream>>>(as2, ad2, src, dst, emax2, E, 1);
    edge_exp<<<blocks((long long)E, T), T, 0, stream>>>(as2, ad2, src, dst, emax2, eexp2, den2, E, 1);
    edge_scatter<<<blocks((long long)E * OUT, T), T, 0, stream>>>(g2, eexp2, den2, src, dst, agg2, E, 1, OUT);

    bias_out<<<blocks((long long)N * OUT, T), T, 0, stream>>>(agg2, b2, (float*)d_out, N * OUT, OUT);
}
